// EGNN_29162827940741
// MI455X (gfx1250) — matrix operand order repaired and checked by reading, not compiled, not run on hardware
//
#include <hip/hip_runtime.h>
#include <hip/hip_bf16.h>

// ---------------- problem constants ----------------
#define BB 2
#define NN 4096
#define DD 128
#define KK 32
#define MD 16
#define HE 256
#define H2 256            // node hidden = 2*D
#define KN1 160           // 144 padded to 160 (5 k-steps of 32)
#define BIGV 1e5f

typedef __attribute__((ext_vector_type(16))) _Float16 v16h;
typedef __attribute__((ext_vector_type(8)))  _Float16 v8h;
typedef __attribute__((ext_vector_type(8)))  float    v8f;

// ---------------- workspace layout (bytes) ----------------
// feats16 : B*N*D f16                      = 2,097,152
// w_e1T   : [256][256] f16                 =   131,072
// w_last  : [256] f32 (rel-dist row)       =     1,024
// w_e2T   : [16][256] f16                  =     8,192
// w_n1T   : [256][160] f16 (zero-padded)   =    81,920
// w_n2T   : [128][256] f16                 =    65,536
// idx     : B*N*K i32                      = 1,048,576
// relk    : B*N*K f32                      = 1,048,576
// emask   : B*N*K f32                      = 1,048,576
// m_i     : B*N*16 f32                     =   524,288
// total ~ 6.06 MB
static constexpr size_t OFF_FEATS16 = 0;
static constexpr size_t OFF_WE1T    = 2097152;
static constexpr size_t OFF_WLAST   = 2228224;
static constexpr size_t OFF_WE2T    = 2229248;
static constexpr size_t OFF_WN1T    = 2237440;
static constexpr size_t OFF_WN2T    = 2319360;
static constexpr size_t OFF_IDX     = 2384896;
static constexpr size_t OFF_RELK    = 3433472;
static constexpr size_t OFF_EMASK   = 4482048;
static constexpr size_t OFF_MI      = 5530624;

// ---------------- helpers ----------------
__device__ __forceinline__ float silu_f(float x) {
    return x / (1.0f + __expf(-x));
}

__device__ __forceinline__ v8f wmma16(v16h a, v16h b, v8f c) {
    return __builtin_amdgcn_wmma_f32_16x16x32_f16(
        /*neg_a=*/false, a, /*neg_b=*/false, b,
        /*c_mod=*/(short)0, c, /*reuse_a=*/false, /*reuse_b=*/false);
}

// A-fragment (16x32 f16) from LDS, row-major, row stride `ld` halves.
// tile = &A[m0*ld + k0]. Lane<16: row=lane, K=k0..k0+15;
// lane>=16: row=lane-16, K=k0+16..k0+31.
__device__ __forceinline__ v16h load_a_lds(const _Float16* tile, int ld, int lane) {
    int hf = (lane >> 4) & 1;
    int r  = lane & 15;
    const _Float16* p = tile + r * ld + hf * 16;
    v8h lo = *(const v8h*)(p);
    v8h hi = *(const v8h*)(p + 8);
    return __builtin_shufflevector(lo, hi, 0,1,2,3,4,5,6,7,8,9,10,11,12,13,14,15);
}

// B-fragment (32x16 f16) from global transposed weights wT[col][k], k stride ldk.
// Lane<16: col, K=k0..k0+15; lane>=16: col, K=k0+16..k0+31 (col passed by caller).
__device__ __forceinline__ v16h load_b_gl(const _Float16* wT, int ldk, int col,
                                          int k0, int lane) {
    int hf = (lane >> 4) & 1;
    return *(const v16h*)(wT + (size_t)col * ldk + k0 + hf * 16);
}

// ---------------- kernel 0: weight / feature prep ----------------
__global__ __launch_bounds__(256)
void egnn_prep(const float* __restrict__ feats,
               const float* __restrict__ w_e1, const float* __restrict__ w_e2,
               const float* __restrict__ w_n1, const float* __restrict__ w_n2,
               _Float16* __restrict__ feats16,
               _Float16* __restrict__ w_e1T, float* __restrict__ w_last,
               _Float16* __restrict__ w_e2T,
               _Float16* __restrict__ w_n1T, _Float16* __restrict__ w_n2T) {
    int tid = blockIdx.x * 256 + threadIdx.x;   // grid covers 1,048,576
    // feats -> f16
    feats16[tid] = (_Float16)feats[tid];
    // w_e1T [256 cols][256 k]  (w_e1 is [257][256] row-major)
    if (tid < 65536) {
        int n = tid >> 8, k = tid & 255;
        w_e1T[n * 256 + k] = (_Float16)w_e1[k * HE + n];
    }
    if (tid < 256) w_last[tid] = w_e1[256 * HE + tid];
    // w_e2T [16][256]  (w_e2 is [256][16])
    if (tid < 4096) {
        int n = tid >> 8, k = tid & 255;
        w_e2T[n * 256 + k] = (_Float16)w_e2[k * MD + n];
    }
    // w_n1T [256][160]  (w_n1 is [144][256]); zero-pad k>=144
    if (tid < 256 * KN1) {
        int n = tid / KN1, k = tid % KN1;
        w_n1T[n * KN1 + k] = (k < DD + MD) ? (_Float16)w_n1[k * H2 + n]
                                           : (_Float16)0.0f;
    }
    // w_n2T [128][256]  (w_n2 is [256][128])
    if (tid < 128 * 256) {
        int n = tid >> 8, k = tid & 255;
        w_n2T[n * 256 + k] = (_Float16)w_n2[k * DD + n];
    }
}

// ---------------- kernel 1: top-K neighbor selection ----------------
__global__ __launch_bounds__(256)
void egnn_topk(const float* __restrict__ coors,
               const unsigned char* __restrict__ mask,
               const unsigned char* __restrict__ adj,
               int* __restrict__ idx_out, float* __restrict__ relk_out,
               float* __restrict__ emask_out) {
    __shared__ float rank_s[NN];
    __shared__ unsigned long long red64[256];
    const int node = blockIdx.x;           // b*N + i
    const int b = node / NN, i = node - b * NN;
    const int t = threadIdx.x;
    const float cx = coors[(size_t)node * 3 + 0];
    const float cy = coors[(size_t)node * 3 + 1];
    const float cz = coors[(size_t)node * 3 + 2];
    const bool  mi = mask[node] != 0;
    const unsigned char* adj_row = adj + (size_t)node * NN;
    const size_t cbase = (size_t)b * NN;

    for (int j = t; j < NN; j += 256) {
        float dx = cx - coors[(cbase + j) * 3 + 0];
        float dy = cy - coors[(cbase + j) * 3 + 1];
        float dz = cz - coors[(cbase + j) * 3 + 2];
        float d = dx * dx + dy * dy + dz * dz;
        bool mj = mask[cbase + j] != 0;
        float rnk = (mi && mj) ? d : BIGV;
        if (j == i)               rnk = -1.0f;
        else if (adj_row[j] != 0) rnk = 0.0f;
        rank_s[j] = rnk;
    }
    __syncthreads();

    for (int kk = 0; kk < KK; ++kk) {
        unsigned long long best = ~0ull;
        for (int j = t; j < NN; j += 256) {
            unsigned int bits = __float_as_uint(rank_s[j]);
            bits = (bits & 0x80000000u) ? ~bits : (bits | 0x80000000u);
            unsigned long long key = ((unsigned long long)bits << 32) | (unsigned)j;
            if (key < best) best = key;
        }
        red64[t] = best;
        __syncthreads();
        for (int s = 128; s > 0; s >>= 1) {
            if (t < s) {
                unsigned long long o = red64[t + s];
                if (o < red64[t]) red64[t] = o;
            }
            __syncthreads();
        }
        if (t == 0) {
            int jstar = (int)(red64[0] & 0xffffffffull);
            float rv = rank_s[jstar];
            float dx = cx - coors[(cbase + jstar) * 3 + 0];
            float dy = cy - coors[(cbase + jstar) * 3 + 1];
            float dz = cz - coors[(cbase + jstar) * 3 + 2];
            float d = dx * dx + dy * dy + dz * dz;
            size_t o = (size_t)node * KK + kk;
            idx_out[o]  = jstar;
            relk_out[o] = d;
            bool mj = mask[cbase + jstar] != 0;
            emask_out[o] = (mi && mj && (rv <= 0.0f)) ? 1.0f : 0.0f;
            rank_s[jstar] = 3.0e38f;      // remove from candidates
        }
        __syncthreads();
    }
}

// ---------------- kernel 2: edge MLP + masked aggregation (WMMA) ----------------
#define AS 264     // 256 + 8 halves pad (16B row stagger -> conflict-free b128)

__global__ __launch_bounds__(256)
void egnn_edge(const _Float16* __restrict__ feats16,
               const int* __restrict__ idx, const float* __restrict__ relk,
               const float* __restrict__ emask,
               const _Float16* __restrict__ w_e1T, const float* __restrict__ w_last,
               const float* __restrict__ b_e1,
               const _Float16* __restrict__ w_e2T, const float* __restrict__ b_e2,
               float* __restrict__ m_i) {
    __shared__ alignas(32) _Float16 Ael[32 * AS];   // edge_in f16 (first 256 feats)
    __shared__ alignas(32) _Float16 H1[32 * AS];    // silu(layer1) f16
    __shared__ float red[32 * 16];
    __shared__ float s_rel[32], s_msk[32];
    __shared__ int   s_idx[32];

    const int node = blockIdx.x;                    // b*N + i
    const int b = node / NN;
    const int t = threadIdx.x;
    const int wv = t >> 5, lane = t & 31;
    const int lc = lane & 15, hf = (lane >> 4) & 1;

    if (t < KK) {
        size_t o = (size_t)node * KK + t;
        s_idx[t] = idx[o];
        s_rel[t] = relk[o];
        s_msk[t] = emask[o];
    }
    __syncthreads();

    // gather edge_in rows: [feats_i | feats_j] (f16, 256 halves)
    {
        int e = t >> 3;                 // edge 0..31
        int c = (t & 7) * 16;           // 16-half chunk
        const v16h* src_i = (const v16h*)(feats16 + (size_t)node * DD + c);
        const v16h* src_j = (const v16h*)(feats16 + ((size_t)b * NN + s_idx[e]) * DD + c);
        *(v16h*)(Ael + e * AS + c)       = *src_i;
        *(v16h*)(Ael + e * AS + 128 + c) = *src_j;
    }
    __syncthreads();

    // layer 1: [32,256] x [256,256]; wave w -> n-tiles {2w,2w+1}, both m-tiles
    for (int nt = 2 * wv; nt <= 2 * wv + 1; ++nt) {
        int n0 = nt * 16, col = n0 + lc;
        v8f acc0 = {}, acc1 = {};
        for (int ks = 0; ks < 8; ++ks) {
            int k0 = ks * 32;
            v16h bf = load_b_gl(w_e1T, 256, col, k0, lane);
            v16h a0 = load_a_lds(Ael + k0, AS, lane);
            v16h a1 = load_a_lds(Ael + 16 * AS + k0, AS, lane);
            acc0 = wmma16(a0, bf, acc0);
            acc1 = wmma16(a1, bf, acc1);
        }
        float bias = b_e1[col], wl = w_last[col];
#pragma unroll
        for (int r = 0; r < 8; ++r) {
            int row = hf * 8 + r;
            H1[row * AS + col] = (_Float16)silu_f(acc0[r] + bias + s_rel[row] * wl);
            int row2 = 16 + row;
            H1[row2 * AS + col] = (_Float16)silu_f(acc1[r] + bias + s_rel[row2] * wl);
        }
    }
    __syncthreads();

    // layer 2: [32,256] x [256,16]; waves 0/1 take m-tiles 0/1
    if (wv < 2) {
        int m0 = wv * 16;
        v8f acc = {};
        for (int ks = 0; ks < 8; ++ks) {
            v16h bf = load_b_gl(w_e2T, 256, lc, ks * 32, lane);
            v16h a  = load_a_lds(H1 + m0 * AS + ks * 32, AS, lane);
            acc = wmma16(a, bf, acc);
        }
        float bias = b_e2[lc];
#pragma unroll
        for (int r = 0; r < 8; ++r) {
            int row = m0 + hf * 8 + r;
            red[row * 16 + lc] = silu_f(acc[r] + bias) * s_msk[row];
        }
    }
    __syncthreads();

    // masked sum over K edges -> m_i[16]
    if (t < MD) {
        float s = 0.0f;
        for (int r = 0; r < KK; ++r) s += red[r * 16 + t];
        m_i[(size_t)node * MD + t] = s;
    }
}

// ---------------- kernel 3: node MLP + residual (WMMA) ----------------
#define HS 168     // 160 + 8 halves pad

__global__ __launch_bounds__(256)
void egnn_node(const _Float16* __restrict__ feats16,
               const float* __restrict__ feats, const float* __restrict__ m_i,
               const _Float16* __restrict__ w_n1T, const float* __restrict__ b_n1,
               const _Float16* __restrict__ w_n2T, const float* __restrict__ b_n2,
               float* __restrict__ out) {
    __shared__ alignas(32) _Float16 Hh[32 * HS];    // [feats | m_i | 0-pad]
    __shared__ alignas(32) _Float16 H1[32 * AS];

    const int node0 = blockIdx.x * 32;
    const int t = threadIdx.x;
    const int wv = t >> 5, lane = t & 31;
    const int lc = lane & 15, hf = (lane >> 4) & 1;

    {   // feats part (128 halves)
        int e = t >> 3, c = (t & 7) * 16;
        *(v16h*)(Hh + e * HS + c) =
            *(const v16h*)(feats16 + (size_t)(node0 + e) * DD + c);
    }
    if (t < 32) {   // m_i (16) + zero pad (16)
        for (int q = 0; q < MD; ++q)
            Hh[t * HS + DD + q] = (_Float16)m_i[(size_t)(node0 + t) * MD + q];
        for (int q = 0; q < 16; ++q)
            Hh[t * HS + DD + MD + q] = (_Float16)0.0f;
    }
    __syncthreads();

    // layer 1: [32,160] x [160,256]; wave w -> n-tiles {2w,2w+1}
    for (int nt = 2 * wv; nt <= 2 * wv + 1; ++nt) {
        int n0 = nt * 16, col = n0 + lc;
        v8f acc0 = {}, acc1 = {};
        for (int ks = 0; ks < 5; ++ks) {
            int k0 = ks * 32;
            v16h bf = load_b_gl(w_n1T, KN1, col, k0, lane);
            v16h a0 = load_a_lds(Hh + k0, HS, lane);
            v16h a1 = load_a_lds(Hh + 16 * HS + k0, HS, lane);
            acc0 = wmma16(a0, bf, acc0);
            acc1 = wmma16(a1, bf, acc1);
        }
        float bias = b_n1[col];
#pragma unroll
        for (int r = 0; r < 8; ++r) {
            int row = hf * 8 + r;
            H1[row * AS + col]        = (_Float16)silu_f(acc0[r] + bias);
            H1[(16 + row) * AS + col] = (_Float16)silu_f(acc1[r] + bias);
        }
    }
    __syncthreads();

    // layer 2: [32,256] x [256,128]; wave w -> n-tile w; + bias + residual
    {
        int col = wv * 16 + lc;
        v8f acc0 = {}, acc1 = {};
        for (int ks = 0; ks < 8; ++ks) {
            int k0 = ks * 32;
            v16h bf = load_b_gl(w_n2T, 256, col, k0, lane);
            v16h a0 = load_a_lds(H1 + k0, AS, lane);
            v16h a1 = load_a_lds(H1 + 16 * AS + k0, AS, lane);
            acc0 = wmma16(a0, bf, acc0);
            acc1 = wmma16(a1, bf, acc1);
        }
        float bias = b_n2[col];
#pragma unroll
        for (int r = 0; r < 8; ++r) {
            int row = hf * 8 + r;
            size_t o0 = (size_t)(node0 + row) * DD + col;
            out[o0] = acc0[r] + bias + feats[o0];
            size_t o1 = (size_t)(node0 + 16 + row) * DD + col;
            out[o1] = acc1[r] + bias + feats[o1];
        }
    }
}

// ---------------- kernel 4: coors pass-through into output tail ----------------
__global__ __launch_bounds__(256)
void egnn_coors(const float* __restrict__ coors, float* __restrict__ out_tail) {
    int tid = blockIdx.x * 256 + threadIdx.x;
    if (tid < BB * NN * 3) out_tail[tid] = coors[tid];
}

// ---------------- launcher ----------------
extern "C" void kernel_launch(void* const* d_in, const int* in_sizes, int n_in,
                              void* d_out, int out_size, void* d_ws, size_t ws_size,
                              hipStream_t stream) {
    (void)in_sizes; (void)n_in; (void)out_size; (void)ws_size;
    const float*         feats = (const float*)d_in[0];
    const float*         coors = (const float*)d_in[1];
    const unsigned char* mask  = (const unsigned char*)d_in[2];
    const unsigned char* adj   = (const unsigned char*)d_in[3];
    // d_in[4] = num_nearest (compile-time KK here)
    const float* w_e1 = (const float*)d_in[5];
    const float* b_e1 = (const float*)d_in[6];
    const float* w_e2 = (const float*)d_in[7];
    const float* b_e2 = (const float*)d_in[8];
    const float* w_n1 = (const float*)d_in[9];
    const float* b_n1 = (const float*)d_in[10];
    const float* w_n2 = (const float*)d_in[11];
    const float* b_n2 = (const float*)d_in[12];

    char* ws = (char*)d_ws;
    _Float16* feats16 = (_Float16*)(ws + OFF_FEATS16);
    _Float16* w_e1T   = (_Float16*)(ws + OFF_WE1T);
    float*    w_last  = (float*)   (ws + OFF_WLAST);
    _Float16* w_e2T   = (_Float16*)(ws + OFF_WE2T);
    _Float16* w_n1T   = (_Float16*)(ws + OFF_WN1T);
    _Float16* w_n2T   = (_Float16*)(ws + OFF_WN2T);
    int*      idxb    = (int*)     (ws + OFF_IDX);
    float*    relk    = (float*)   (ws + OFF_RELK);
    float*    emask   = (float*)   (ws + OFF_EMASK);
    float*    mi      = (float*)   (ws + OFF_MI);

    float* out = (float*)d_out;

    egnn_prep<<<4096, 256, 0, stream>>>(feats, w_e1, w_e2, w_n1, w_n2,
                                        feats16, w_e1T, w_last, w_e2T, w_n1T, w_n2T);
    egnn_topk<<<BB * NN, 256, 0, stream>>>(coors, mask, adj, idxb, relk, emask);
    egnn_edge<<<BB * NN, 256, 0, stream>>>(feats16, idxb, relk, emask,
                                           w_e1T, w_last, b_e1, w_e2T, b_e2, mi);
    egnn_node<<<(BB * NN) / 32, 256, 0, stream>>>(feats16, feats, mi,
                                                  w_n1T, b_n1, w_n2T, b_n2, out);
    egnn_coors<<<(BB * NN * 3 + 255) / 256, 256, 0, stream>>>(
        coors, out + (size_t)BB * NN * DD);
}
